// DecoderLayerCell_25168508354864
// MI455X (gfx1250) — matrix operand order repaired and checked by reading, not compiled, not run on hardware
//
#include <hip/hip_runtime.h>

// ---------------------------------------------------------------------------
// CDNA5 (gfx1250) decoder layer: f16 WMMA GEMMs (TDM-staged B operand,
// double-buffered LDS), WMMA attention, fp32 epilogues.
// ---------------------------------------------------------------------------

typedef __attribute__((ext_vector_type(16))) _Float16 v16h;
typedef __attribute__((ext_vector_type(8)))  _Float16 v8h;
typedef __attribute__((ext_vector_type(8)))  float    v8f;
typedef __attribute__((ext_vector_type(4)))  unsigned int v4u;
typedef __attribute__((ext_vector_type(8)))  int      v8i;
typedef __attribute__((ext_vector_type(4)))  int      v4i;

union HFrag { v16h v; v8h h[2]; _Float16 e[16]; };

constexpr int Bc = 8, Tc = 1024, Dc = 512, Hc = 8, WINc = 64, Sc = 800, DFFc = 2048, DKc = 64;
constexpr int Mr = Bc * Tc;            // 8192 token rows
constexpr float NEGV = -65500.0f;

#if defined(__HIP_DEVICE_COMPILE__) && __has_builtin(__builtin_amdgcn_tensor_load_to_lds) && \
    __has_builtin(__builtin_amdgcn_s_wait_tensorcnt)
#define HAVE_TDM 1
#else
#define HAVE_TDM 0
#endif

// A-matrix fragment (16x32 f16): lane holds row (lane&15); K chunks hs*8..+7 and 16+hs*8..+7
__device__ __forceinline__ v16h frag_a(const _Float16* p) {
  HFrag f;
  f.h[0] = *(const v8h*)(p);
  f.h[1] = *(const v8h*)(p + 16);
  return f.v;
}
// B-matrix fragment (32x16 f16) from [N][K] storage: lane holds col (lane&15); K chunks hs*8..+7 and 16+hs*8..+7
__device__ __forceinline__ v16h frag_b(const _Float16* p) {
  HFrag f;
  f.h[0] = *(const v8h*)(p);
  f.h[1] = *(const v8h*)(p + 16);
  return f.v;
}
// A fragment assembled from fp32 data (softmax probabilities in LDS)
__device__ __forceinline__ v16h frag_a_f32(const float* p) {
  HFrag f;
#pragma unroll
  for (int i = 0; i < 8; ++i) { f.e[i] = (_Float16)p[i]; f.e[8 + i] = (_Float16)p[16 + i]; }
  return f.v;
}

#define WMMA_F16(a, b, c) \
  __builtin_amdgcn_wmma_f32_16x16x32_f16(false, (a), false, (b), (short)0, (c), false, false)

// ---------------------------------------------------------------------------
// Tensor Data Mover: DMA a 2D f16 tile (rows x 32 elems) from global [N][K]
// into LDS (packed row-major, pitch 32). Descriptor per CDNA5 ISA section 8.
// Issued by one wave; completion via TENSORcnt.
// ---------------------------------------------------------------------------
__device__ __forceinline__ void tdm_load_tile(const _Float16* src, _Float16* ldsDst,
                                              unsigned tensorK, unsigned tensorN) {
#if HAVE_TDM
  const unsigned long long ga = (unsigned long long)(size_t)src;
  const unsigned ldsOff = (unsigned)(size_t)(void*)ldsDst;  // LDS aperture: offset = addr[31:0]
  v4u g0;
  g0.x = 1u;                                            // count=1, user descriptor
  g0.y = ldsOff;                                        // lds_addr (bytes)
  g0.z = (unsigned)(ga & 0xFFFFFFFFu);                  // global_addr[31:0]
  g0.w = (unsigned)((ga >> 32) & 0x1FFFFFFu) | (2u << 30);  // global_addr[56:32] | type=2
  v8i g1;
  g1[0] = (int)(1u << 16);                              // workgroup_mask=0, data_size=2B
  g1[1] = (int)((tensorK & 0xFFFFu) << 16);             // tensor_dim0[15:0]
  g1[2] = (int)((tensorK >> 16) | ((tensorN & 0xFFFFu) << 16));  // td0 hi | td1 lo
  g1[3] = (int)((tensorN >> 16) | (32u << 16));         // td1 hi | tile_dim0 = 32
  g1[4] = 128;                                          // tile_dim1 = 128 rows, tile_dim2 = 0
  g1[5] = (int)tensorK;                                 // tensor_dim0_stride[31:0]
  g1[6] = 0;                                            // stride hi, dim1_stride lo (unused, 2D)
  g1[7] = 0;
  v4i z = {0, 0, 0, 0};
#if defined(__clang_major__) && (__clang_major__ >= 23)
  v8i z8 = {0, 0, 0, 0, 0, 0, 0, 0};
  __builtin_amdgcn_tensor_load_to_lds(g0, g1, z, z, z8, 0);
#else
  __builtin_amdgcn_tensor_load_to_lds(g0, g1, z, z, 0);
#endif
#endif
}

// Stage one 128x32 B tile (may overlap with compute); wait separately.
__device__ __forceinline__ void stage_b(const _Float16* src, _Float16* dst, int K, int N,
                                        int tidx) {
#if HAVE_TDM
  if (tidx < 32) tdm_load_tile(src, dst, (unsigned)K, (unsigned)N);
#else
  for (int idx = tidx; idx < 512; idx += 256) {     // 128 rows x 32 halves, 8 halves/iter
    const int r = idx >> 2, c = (idx & 3) * 8;
    *(v8h*)(dst + r * 32 + c) = *(const v8h*)(src + (size_t)r * K + c);
  }
#endif
}
__device__ __forceinline__ void stage_b_wait(int tidx) {
#if HAVE_TDM
  if (tidx < 32) __builtin_amdgcn_s_wait_tensorcnt(0);
#endif
}

// ---------------------------------------------------------------------------
// LayerNorm over D=512 -> f16 output (one block of 128 threads per row)
// ---------------------------------------------------------------------------
__global__ void ln_to_f16(const float* __restrict__ x, const float* __restrict__ g,
                          const float* __restrict__ be, _Float16* __restrict__ out) {
  __shared__ float r1[128], r2[128];
  const int row = blockIdx.x, tid = threadIdx.x;
  const float* px = x + (size_t)row * Dc;
  float v[4], s = 0.f, s2 = 0.f;
#pragma unroll
  for (int i = 0; i < 4; ++i) { v[i] = px[tid + i * 128]; s += v[i]; s2 += v[i] * v[i]; }
  r1[tid] = s; r2[tid] = s2; __syncthreads();
  for (int st = 64; st > 0; st >>= 1) {
    if (tid < st) { r1[tid] += r1[tid + st]; r2[tid] += r2[tid + st]; }
    __syncthreads();
  }
  const float mean = r1[0] * (1.f / Dc);
  const float var  = r2[0] * (1.f / Dc) - mean * mean;
  const float rs = rsqrtf(var + 1e-5f);
  _Float16* po = out + (size_t)row * Dc;
#pragma unroll
  for (int i = 0; i < 4; ++i) {
    const int c = tid + i * 128;
    po[c] = (_Float16)((v[i] - mean) * rs * g[c] + be[c]);
  }
}

// cast fp32 weight (K x N, row-major) -> f16 transposed (N x K) for B fragments
__global__ void cast_wt(const float* __restrict__ W, _Float16* __restrict__ Wt, int K, int N) {
  const int idx = blockIdx.x * 256 + threadIdx.x;
  if (idx < K * N) {
    const int nn = idx / K, kk = idx - nn * K;
    Wt[idx] = (_Float16)W[(size_t)kk * N + nn];
  }
}

__global__ void cast_f16v(const float* __restrict__ in, _Float16* __restrict__ out, int n) {
  const int idx = blockIdx.x * 256 + threadIdx.x;
  if (idx < n) out[idx] = (_Float16)in[idx];
}

// vf (Sc x Dc) f16 -> vt (Dc x Sc) f16 so PV B-fragments are contiguous in K=s
__global__ void transpose_f16(const _Float16* __restrict__ in, _Float16* __restrict__ out) {
  const int idx = blockIdx.x * 256 + threadIdx.x;
  if (idx < Sc * Dc) {
    const int d = idx / Sc, s = idx - d * Sc;
    out[idx] = in[(size_t)s * Dc + d];
  }
}

// ---------------------------------------------------------------------------
// WMMA GEMM: C(MxN) = A(MxK) f16 @ Wt(NxK)^T f16 + bias.
// Block = 256 threads (8 waves); block tile 128x128; wave tile 32x64.
// B operand staged per 32-K chunk by the TDM into double-buffered LDS.
// EPI: 0 = bias only, 1 = bias+ReLU, 2 = bias+residual add.
// ---------------------------------------------------------------------------
template <int EPI>
__global__ __launch_bounds__(256) void gemm_wmma(
    const _Float16* __restrict__ A, const _Float16* __restrict__ Wt,
    const float* __restrict__ bias, const float* __restrict__ resid,
    float* __restrict__ Cf, _Float16* __restrict__ Ch, int M, int N, int K) {
  __shared__ _Float16 bstage[2][128 * 32];     // 2 x 8KB B tiles: [col][k] pitch 32
  const int lane = threadIdx.x & 31, wave = threadIdx.x >> 5;
  const int lr = lane & 15, hs = lane >> 4;
  const int wr = wave >> 1;                    // row group (32 rows each)
  const int wc = wave & 1;                     // col group (64 cols each)
  const int col0 = blockIdx.x * 128;
  const int row0 = blockIdx.y * 128 + wr * 32;
  int ar0 = row0 + lr;      ar0 = ar0 < M ? ar0 : M - 1;   // clamp; stores guarded
  int ar1 = row0 + 16 + lr; ar1 = ar1 < M ? ar1 : M - 1;
  const _Float16* pa0 = A + (size_t)ar0 * K + hs * 8;
  const _Float16* pa1 = A + (size_t)ar1 * K + hs * 8;
  const _Float16* pwt = Wt + (size_t)col0 * K;

  v8f acc[2][4] = {};
  const int NK = K / 32;
  stage_b(pwt, bstage[0], K, N, threadIdx.x);
  stage_b_wait(threadIdx.x);
  __syncthreads();
  for (int kc = 0; kc < NK; ++kc) {
    if (kc + 1 < NK)                              // prefetch next chunk into alternate buffer
      stage_b(pwt + (size_t)(kc + 1) * 32, bstage[(kc + 1) & 1], K, N, threadIdx.x);
    const _Float16* bs = bstage[kc & 1] + (wc * 64 + lr) * 32 + hs * 8;
    v16h a0 = frag_a(pa0 + kc * 32);
    v16h a1 = frag_a(pa1 + kc * 32);
    __builtin_prefetch(pa0 + kc * 32 + 512, 0, 3);          // global_prefetch_b8
#pragma unroll
    for (int ct = 0; ct < 4; ++ct) {
      v16h bf = frag_b(bs + ct * 16 * 32);
      acc[0][ct] = WMMA_F16(a0, bf, acc[0][ct]);
      acc[1][ct] = WMMA_F16(a1, bf, acc[1][ct]);
    }
    stage_b_wait(threadIdx.x);
    __syncthreads();
  }
  const int mb = hs * 8;
#pragma unroll
  for (int rt = 0; rt < 2; ++rt) {
#pragma unroll
    for (int ct = 0; ct < 4; ++ct) {
      const int n = col0 + wc * 64 + ct * 16 + lr;
      const float bv = bias[n];
#pragma unroll
      for (int i = 0; i < 8; ++i) {
        const int m = row0 + rt * 16 + mb + i;
        if (m < M) {
          float val = acc[rt][ct][i] + bv;
          if (EPI == 1) val = fmaxf(val, 0.f);
          if (EPI == 2) val += resid[(size_t)m * N + n];
          if (Cf) Cf[(size_t)m * N + n] = val;
          if (Ch) Ch[(size_t)m * N + n] = (_Float16)val;
        }
      }
    }
  }
}

// ---------------------------------------------------------------------------
// Local causal sliding-window attention, one block per (window, head, batch).
// 64 queries x 128 keys (prev window + current), DK = 64. 4 waves.
// ---------------------------------------------------------------------------
__global__ void local_attn(const _Float16* __restrict__ q16, const _Float16* __restrict__ k16,
                           const _Float16* __restrict__ v16, const int* __restrict__ tmask,
                           _Float16* __restrict__ out16) {
  __shared__ float sc[64][132];          // scores, fp32 (padded pitch)
  __shared__ _Float16 vt[64][128];       // v2 transposed: [dk][key]
  const int n = blockIdx.x, h = blockIdx.y, b = blockIdx.z;
  const int lane = threadIdx.x & 31, wave = threadIdx.x >> 5;
  const int lr = lane & 15, hs = lane >> 4;
  const int t0 = n * WINc;
  const size_t rowbase = (size_t)b * Tc;

  // ---- scores = q @ k2^T (wave computes 16 rows x 128 keys = 8 tiles) ----
  const _Float16* pq = q16 + (rowbase + t0 + wave * 16 + lr) * Dc + h * DKc;
  v16h a0 = frag_a(pq + hs * 8);
  v16h a1 = frag_a(pq + 32 + hs * 8);
  v8f acc[8] = {};
#pragma unroll
  for (int ct = 0; ct < 8; ++ct) {
    const bool valid = (n > 0) || (ct >= 4);     // wave-uniform: first window's prev half is pad
    v16h b0 = {}, b1 = {};
    if (valid) {
      const _Float16* pk = k16 + (rowbase + (t0 - WINc + ct * 16 + lr)) * Dc + h * DKc;
      b0 = frag_b(pk + hs * 8);
      b1 = frag_b(pk + 32 + hs * 8);
    }
    acc[ct] = WMMA_F16(a0, b0, acc[ct]);
    acc[ct] = WMMA_F16(a1, b1, acc[ct]);
  }
  // scale + mask -> LDS
#pragma unroll
  for (int ct = 0; ct < 8; ++ct) {
    const int j = ct * 16 + lr;
    const int kpos = t0 - WINc + j;
    const bool kok = (kpos >= 0) && (tmask[b * Tc + kpos] != 0);
#pragma unroll
    for (int i = 0; i < 8; ++i) {
      const int m = wave * 16 + hs * 8 + i;                     // local query row
      const bool bad = !kok || ((j >= WINc) && (j - WINc > m)); // causal: key pos > query pos
      sc[m][j] = bad ? NEGV : acc[ct][i] * 0.125f;              // *(DK^-0.5)
    }
  }
  // stage v2^T into LDS
  for (int idx = threadIdx.x; idx < 64 * 128; idx += 128) {
    const int d = idx >> 7, j = idx & 127;
    const int kpos = t0 - WINc + j;
    vt[d][j] = (kpos >= 0) ? v16[(rowbase + kpos) * Dc + h * DKc + d] : (_Float16)0.f;
  }
  __syncthreads();
  // ---- row softmax (threads 0..63, one row each) ----
  if (threadIdx.x < 64) {
    const int m = threadIdx.x;
    float mx = -3.0e38f;
    for (int j = 0; j < 128; ++j) mx = fmaxf(mx, sc[m][j]);
    float sum = 0.f;
    for (int j = 0; j < 128; ++j) { float e = __expf(sc[m][j] - mx); sc[m][j] = e; sum += e; }
    const float inv = 1.f / sum;
    for (int j = 0; j < 128; ++j) sc[m][j] *= inv;
  }
  __syncthreads();
  // ---- out = attn @ v2 (wave: 16 rows x 64 dk cols, K = 128) ----
  v8f o[4] = {};
#pragma unroll
  for (int k0 = 0; k0 < 128; k0 += 32) {
    v16h a = frag_a_f32(&sc[wave * 16 + lr][k0 + hs * 8]);
#pragma unroll
    for (int ct = 0; ct < 4; ++ct) {
      v16h bf = frag_b(&vt[ct * 16 + lr][k0 + hs * 8]);
      o[ct] = WMMA_F16(a, bf, o[ct]);
    }
  }
#pragma unroll
  for (int ct = 0; ct < 4; ++ct) {
    const int dcol = ct * 16 + lr;
#pragma unroll
    for (int i = 0; i < 8; ++i) {
      const int m = wave * 16 + hs * 8 + i;
      out16[(rowbase + t0 + m) * Dc + h * DKc + dcol] = (_Float16)o[ct][i];
    }
  }
}

// ---------------------------------------------------------------------------
// Global cross-attention vs S=800 feature keys. Block = 16 query rows, 4 waves.
// ---------------------------------------------------------------------------
__global__ void global_attn(const _Float16* __restrict__ q16, const _Float16* __restrict__ kf16,
                            const _Float16* __restrict__ vt16, const int* __restrict__ smask,
                            _Float16* __restrict__ out16) {
  __shared__ float sc[16][804];          // 16 x 800 scores (padded pitch), ~51KB
  const int h = blockIdx.y, b = blockIdx.z;
  const int lane = threadIdx.x & 31, wave = threadIdx.x >> 5;
  const int lr = lane & 15, hs = lane >> 4;
  const int t0 = blockIdx.x * 16;
  const size_t rowbase = (size_t)b * Tc;

  // ---- scores = q @ kf^T; waves split the 50 column tiles ----
  const _Float16* pq = q16 + (rowbase + t0 + lr) * Dc + h * DKc;
  v16h a0 = frag_a(pq + hs * 8);
  v16h a1 = frag_a(pq + 32 + hs * 8);
  for (int ct = wave; ct < Sc / 16; ct += 4) {
    v8f acc = {};
    const _Float16* pk = kf16 + (size_t)(ct * 16 + lr) * Dc + h * DKc;
    acc = WMMA_F16(a0, frag_b(pk + hs * 8), acc);
    acc = WMMA_F16(a1, frag_b(pk + 32 + hs * 8), acc);
    const int s = ct * 16 + lr;
    const bool ok = smask[s] != 0;
#pragma unroll
    for (int i = 0; i < 8; ++i) sc[hs * 8 + i][s] = ok ? acc[i] * 0.125f : NEGV;
  }
  __syncthreads();
  // ---- row softmax ----
  if (threadIdx.x < 16) {
    const int m = threadIdx.x;
    float mx = -3.0e38f;
    for (int s = 0; s < Sc; ++s) mx = fmaxf(mx, sc[m][s]);
    float sum = 0.f;
    for (int s = 0; s < Sc; ++s) { float e = __expf(sc[m][s] - mx); sc[m][s] = e; sum += e; }
    const float inv = 1.f / sum;
    for (int s = 0; s < Sc; ++s) sc[m][s] *= inv;
  }
  __syncthreads();
  // ---- out = attn @ vf; wave owns one 16-col dk tile; K = 800 ----
  v8f o = {};
  for (int k0 = 0; k0 < Sc; k0 += 32) {
    v16h a = frag_a_f32(&sc[lr][k0 + hs * 8]);
    v16h bf = frag_b(vt16 + (size_t)(h * DKc + wave * 16 + lr) * Sc + k0 + hs * 8);
    o = WMMA_F16(a, bf, o);
  }
  const int dcol = wave * 16 + lr;
#pragma unroll
  for (int i = 0; i < 8; ++i)
    out16[(rowbase + t0 + hs * 8 + i) * Dc + h * DKc + dcol] = (_Float16)o[i];
}

// ---------------------------------------------------------------------------
// Host orchestration
// ---------------------------------------------------------------------------
extern "C" void kernel_launch(void* const* d_in, const int* in_sizes, int n_in,
                              void* d_out, int out_size, void* d_ws, size_t ws_size,
                              hipStream_t stream) {
  (void)in_sizes; (void)n_in; (void)out_size; (void)ws_size;
  const float* x      = (const float*)d_in[0];
  const float* feat   = (const float*)d_in[1];
  const float* la_wq  = (const float*)d_in[2];  const float* la_bq = (const float*)d_in[3];
  const float* la_wk  = (const float*)d_in[4];  const float* la_bk = (const float*)d_in[5];
  const float* la_wv  = (const float*)d_in[6];  const float* la_bv = (const float*)d_in[7];
  const float* la_wo  = (const float*)d_in[8];  const float* la_bo = (const float*)d_in[9];
  const float* ga_wq  = (const float*)d_in[10]; const float* ga_bq = (const float*)d_in[11];
  const float* ga_wk  = (const float*)d_in[12]; const float* ga_bk = (const float*)d_in[13];
  const float* ga_wv  = (const float*)d_in[14]; const float* ga_bv = (const float*)d_in[15];
  const float* ga_wo  = (const float*)d_in[16]; const float* ga_bo = (const float*)d_in[17];
  const float* ff_w1  = (const float*)d_in[18]; const float* ff_b1 = (const float*)d_in[19];
  const float* ff_w2  = (const float*)d_in[20]; const float* ff_b2 = (const float*)d_in[21];
  const float* ln0_g  = (const float*)d_in[22]; const float* ln0_b = (const float*)d_in[23];
  const float* ln1_g  = (const float*)d_in[24]; const float* ln1_b = (const float*)d_in[25];
  const float* ln2_g  = (const float*)d_in[26]; const float* ln2_b = (const float*)d_in[27];
  const int*   src_mask = (const int*)d_in[28];
  const int*   tgt_mask = (const int*)d_in[29];
  float* out = (float*)d_out;

  char* wsb = (char*)d_ws;
  size_t off = 0;
  auto alloc = [&](size_t bytes) -> void* {
    off = (off + 255) & ~(size_t)255;
    void* p = wsb + off;
    off += bytes;
    return p;
  };
  _Float16* xn16   = (_Float16*)alloc((size_t)Mr * Dc * 2);
  _Float16* q16    = (_Float16*)alloc((size_t)Mr * Dc * 2);
  _Float16* k16    = (_Float16*)alloc((size_t)Mr * Dc * 2);
  _Float16* v16    = (_Float16*)alloc((size_t)Mr * Dc * 2);
  _Float16* att16  = (_Float16*)alloc((size_t)Mr * Dc * 2);
  float*    x1     = (float*)alloc((size_t)Mr * Dc * 4);
  float*    x2     = (float*)alloc((size_t)Mr * Dc * 4);
  _Float16* feat16 = (_Float16*)alloc((size_t)Sc * Dc * 2);
  _Float16* kf16   = (_Float16*)alloc((size_t)Sc * Dc * 2);
  _Float16* vf16   = (_Float16*)alloc((size_t)Sc * Dc * 2);
  _Float16* vt16   = (_Float16*)alloc((size_t)Dc * Sc * 2);
  _Float16* ff16   = (_Float16*)alloc((size_t)Mr * DFFc * 2);
  _Float16* wt     = (_Float16*)alloc((size_t)DFFc * Dc * 2);  // reused sequentially

  const dim3 blk(256);
  const dim3 g512(Dc / 128, Mr / 128);             // N=512, M=8192
  const dim3 gfeat(Dc / 128, (Sc + 127) / 128);    // N=512, M=800
  const int cw = (Dc * Dc + 255) / 256;

  // ---- sublayer 0: windowed causal local self-attention ----
  ln_to_f16<<<Mr, 128, 0, stream>>>(x, ln0_g, ln0_b, xn16);
  cast_wt<<<cw, 256, 0, stream>>>(la_wq, wt, Dc, Dc);
  gemm_wmma<0><<<g512, blk, 0, stream>>>(xn16, wt, la_bq, nullptr, nullptr, q16, Mr, Dc, Dc);
  cast_wt<<<cw, 256, 0, stream>>>(la_wk, wt, Dc, Dc);
  gemm_wmma<0><<<g512, blk, 0, stream>>>(xn16, wt, la_bk, nullptr, nullptr, k16, Mr, Dc, Dc);
  cast_wt<<<cw, 256, 0, stream>>>(la_wv, wt, Dc, Dc);
  gemm_wmma<0><<<g512, blk, 0, stream>>>(xn16, wt, la_bv, nullptr, nullptr, v16, Mr, Dc, Dc);
  local_attn<<<dim3(Tc / WINc, Hc, Bc), 128, 0, stream>>>(q16, k16, v16, tgt_mask, att16);
  cast_wt<<<cw, 256, 0, stream>>>(la_wo, wt, Dc, Dc);
  gemm_wmma<2><<<g512, blk, 0, stream>>>(att16, wt, la_bo, x, x1, nullptr, Mr, Dc, Dc);

  // ---- sublayer 1: global source attention ----
  ln_to_f16<<<Mr, 128, 0, stream>>>(x1, ln1_g, ln1_b, xn16);
  cast_wt<<<cw, 256, 0, stream>>>(ga_wq, wt, Dc, Dc);
  gemm_wmma<0><<<g512, blk, 0, stream>>>(xn16, wt, ga_bq, nullptr, nullptr, q16, Mr, Dc, Dc);
  cast_f16v<<<(Sc * Dc + 255) / 256, 256, 0, stream>>>(feat, feat16, Sc * Dc);
  cast_wt<<<cw, 256, 0, stream>>>(ga_wk, wt, Dc, Dc);
  gemm_wmma<0><<<gfeat, blk, 0, stream>>>(feat16, wt, ga_bk, nullptr, nullptr, kf16, Sc, Dc, Dc);
  cast_wt<<<cw, 256, 0, stream>>>(ga_wv, wt, Dc, Dc);
  gemm_wmma<0><<<gfeat, blk, 0, stream>>>(feat16, wt, ga_bv, nullptr, nullptr, vf16, Sc, Dc, Dc);
  transpose_f16<<<(Sc * Dc + 255) / 256, 256, 0, stream>>>(vf16, vt16);
  global_attn<<<dim3(Tc / 16, Hc, Bc), 128, 0, stream>>>(q16, kf16, vt16, src_mask, att16);
  cast_wt<<<cw, 256, 0, stream>>>(ga_wo, wt, Dc, Dc);
  gemm_wmma<2><<<g512, blk, 0, stream>>>(att16, wt, ga_bo, x1, x2, nullptr, Mr, Dc, Dc);

  // ---- sublayer 2: FFN ----
  ln_to_f16<<<Mr, 128, 0, stream>>>(x2, ln2_g, ln2_b, xn16);
  cast_wt<<<(Dc * DFFc + 255) / 256, 256, 0, stream>>>(ff_w1, wt, Dc, DFFc);
  gemm_wmma<1><<<dim3(DFFc / 128, Mr / 128), blk, 0, stream>>>(xn16, wt, ff_b1, nullptr, nullptr,
                                                               ff16, Mr, DFFc, Dc);
  cast_wt<<<(DFFc * Dc + 255) / 256, 256, 0, stream>>>(ff_w2, wt, DFFc, Dc);
  gemm_wmma<2><<<g512, blk, 0, stream>>>(ff16, wt, ff_b2, x2, out, Mr ? nullptr : nullptr, Mr, Dc, DFFc);
}
